// LSTMClassifier_74071005987503
// MI455X (gfx1250) — matrix operand order repaired and checked by reading, not compiled, not run on hardware
//
#include <hip/hip_runtime.h>

typedef __attribute__((ext_vector_type(16))) _Float16 v16h;
typedef __attribute__((ext_vector_type(8)))  _Float16 v8h;
typedef __attribute__((ext_vector_type(8)))  float    v8f;

#define Bx  256
#define Tx  512
#define Hx  32
#define G4x 128   // 4*H

// v_exp_f32 + v_rcp_f32 based activations (no IEEE div sequences on the
// recurrent critical path).
__device__ __forceinline__ float sigf(float x){
  return __builtin_amdgcn_rcpf(1.0f + __expf(-x));
}
__device__ __forceinline__ float tnhf(float x){
  return 2.0f*__builtin_amdgcn_rcpf(1.0f + __expf(-2.0f*x)) - 1.0f;
}

// A-format (16x32 f16) gather: p points at row base + koff (koff = lane<16?0:8).
// elements 0-7 <- K[koff..koff+7], elements 8-15 <- K[koff+16..koff+23]
__device__ __forceinline__ v16h loadA(const _Float16* p){
  v8h lo = *(const v8h*)p;
  v8h hi = *(const v8h*)(p + 16);
  v16h r;
  #pragma unroll
  for (int e = 0; e < 8; ++e){ r[e] = lo[e]; r[e+8] = hi[e]; }
  return r;
}
// B-format (32x16 f16) gather: p points at column-row base + kbase (kbase = lane<16?0:8).
__device__ __forceinline__ v16h loadB(const _Float16* p){
  v8h lo = *(const v8h*)p;
  v8h hi = *(const v8h*)(p + 16);
  v16h r;
  #pragma unroll
  for (int e = 0; e < 8; ++e){ r[e] = lo[e]; r[e+8] = hi[e]; }
  return r;
}
// Launder a 32-bit index so loads depending on it cannot be hoisted out of the
// T-loop (keeps weight tiles streaming from LDS via ds_load_b128 instead of
// pinning 64-128 VGPRs of loop-invariant data -> avoids scratch spills).
__device__ __forceinline__ int opaque(int v){
  asm volatile("" : "+v"(v));
  return v;
}

// ---------------- Layer 0: bidirectional LSTM over raw input x ----------------
// grid = 2 blocks (dir), 512 threads = 16 waves, each wave = 16 batch rows.
__global__ __launch_bounds__(512)
void lstm_l0(const float* __restrict__ x,
             const float* wih_f, const float* whh_f, const float* bih_f, const float* bhh_f,
             const float* wih_b, const float* whh_b, const float* bih_b, const float* bhh_b,
             _Float16* __restrict__ hs0)
{
  __shared__ _Float16 whh_s[G4x*Hx];     // 8 KB recurrent weights
  __shared__ _Float16 wih_s[G4x*Hx];     // 8 KB input weights, K zero-padded to 32
  __shared__ _Float16 hsl[16][16*Hx];    // per-wave h transpose slabs (16 KB)

  const int dir = blockIdx.x;
  const float* wih = dir ? wih_b : wih_f;
  const float* whh = dir ? whh_b : whh_f;
  const float* bih = dir ? bih_b : bih_f;
  const float* bhh = dir ? bhh_b : bhh_f;

  const int tid = threadIdx.x;
  for (int i = tid; i < G4x*Hx; i += 512){
    whh_s[i] = (_Float16)whh[i];
    int g = i >> 5, k = i & 31;
    wih_s[i] = (k < 3) ? (_Float16)wih[g*3 + k] : (_Float16)0.0f;
  }
  __syncthreads();

  const int wave = tid >> 5, lane = tid & 31;
  const int bt = wave * 16;          // batch tile base
  const int lm = lane & 15;
  const bool hig = lane >= 16;
  _Float16* slab = hsl[wave];

  const int koff  = hig ? 8 : 0;     // A-gather offset
  const int kbase = hig ? 8 : 0;     // B-gather offset
  const int mrow  = hig ? 8 : 0;

  // persistent W_hh B-tiles (recurrent critical path stays in registers)
  v16h whhB[8];
  #pragma unroll
  for (int n = 0; n < 8; ++n)
    whhB[n] = loadB(&whh_s[(n*16 + lm)*Hx + kbase]);

  float biasn[8];
  #pragma unroll
  for (int n = 0; n < 8; ++n){ int g = n*16 + lm; biasn[n] = bih[g] + bhh[g]; }

  v8f cs0, cs1;
  #pragma unroll
  for (int e = 0; e < 8; ++e){ cs0[e] = 0.0f; cs1[e] = 0.0f; }
  v16h ah;                           // A-format h_{t-1}
  #pragma unroll
  for (int e = 0; e < 16; ++e) ah[e] = (_Float16)0.0f;

  for (int t = 0; t < Tx; ++t){
    const int tt = dir ? (Tx-1-t) : t;

    // x tile as K=32-padded A operand (lanes<16: elements 0..2 = K0..2).
    // Branchless: all lanes load (row lm is valid for every lane), hig lanes
    // select 0 via cndmask instead of an exec-mask dance each iteration.
    const float* xp = x + ((size_t)(bt + lm)*Tx + tt)*3;
    float x0 = xp[0], x1 = xp[1], x2 = xp[2];
    v16h ax;
    #pragma unroll
    for (int e = 0; e < 16; ++e) ax[e] = (_Float16)0.0f;
    ax[0] = (_Float16)(hig ? 0.0f : x0);
    ax[1] = (_Float16)(hig ? 0.0f : x1);
    ax[2] = (_Float16)(hig ? 0.0f : x2);

    v8f z[8];
    #pragma unroll
    for (int n = 0; n < 8; ++n){
      v8f c;
      #pragma unroll
      for (int e = 0; e < 8; ++e) c[e] = biasn[n];
      c = __builtin_amdgcn_wmma_f32_16x16x32_f16(false, ah, false, whhB[n], (short)0, c, false, false);
      int bi = opaque((n*16 + lm)*Hx + kbase);       // streamed (non-hoistable) LDS tile
      v16h bx = loadB(&wih_s[bi]);
      c = __builtin_amdgcn_wmma_f32_16x16x32_f16(false, ax, false, bx, (short)0, c, false, false);
      z[n] = c;
    }

    // gates (i,f,g,o = ntile pairs 0/1, 2/3, 4/5, 6/7)
    v8f h0, h1;
    #pragma unroll
    for (int e = 0; e < 8; ++e){
      float i0 = sigf(z[0][e]), f0 = sigf(z[2][e]), g0 = tnhf(z[4][e]), o0 = sigf(z[6][e]);
      cs0[e] = f0*cs0[e] + i0*g0;
      h0[e]  = o0 * tnhf(cs0[e]);
      float i1 = sigf(z[1][e]), f1 = sigf(z[3][e]), g1 = tnhf(z[5][e]), o1 = sigf(z[7][e]);
      cs1[e] = f1*cs1[e] + i1*g1;
      h1[e]  = o1 * tnhf(cs1[e]);
    }

    // scatter h (D layout) to LDS row-major [m][k]
    #pragma unroll
    for (int r = 0; r < 8; ++r){
      slab[(mrow + r)*Hx + lm     ] = (_Float16)h0[r];
      slab[(mrow + r)*Hx + lm + 16] = (_Float16)h1[r];
    }
    // gather A-format h for next step (same-wave DS ops are in-order)
    const _Float16* p = &slab[lm*Hx + koff];
    v8h lo  = *(const v8h*)p;
    v8h hi2 = *(const v8h*)(p + 16);
    #pragma unroll
    for (int e = 0; e < 8; ++e){ ah[e] = lo[e]; ah[e+8] = hi2[e]; }
    // store h_t to global sequence [B,T,2H] f16 (row-major so layer-1 A-loads work)
    _Float16* gp = hs0 + ((size_t)(bt + lm)*Tx + tt)*(2*Hx) + dir*Hx + koff;
    *(v8h*)gp        = lo;
    *(v8h*)(gp + 16) = hi2;
  }
}

// ---------------- Layer 1: bidirectional LSTM over hs0, keep only final h ----------------
__global__ __launch_bounds__(512)
void lstm_l1(const _Float16* __restrict__ hs0,
             const float* wih_f, const float* whh_f, const float* bih_f, const float* bhh_f,
             const float* wih_b, const float* whh_b, const float* bih_b, const float* bhh_b,
             float* __restrict__ hn)
{
  __shared__ _Float16 whh_s[G4x*Hx];      // 8 KB
  __shared__ _Float16 wih_s[G4x*2*Hx];    // 16 KB, [g][k], k<64
  __shared__ _Float16 hsl[16][16*Hx];     // 16 KB

  const int dir = blockIdx.x;
  const float* wih = dir ? wih_b : wih_f;
  const float* whh = dir ? whh_b : whh_f;
  const float* bih = dir ? bih_b : bih_f;
  const float* bhh = dir ? bhh_b : bhh_f;

  const int tid = threadIdx.x;
  for (int i = tid; i < G4x*Hx;   i += 512) whh_s[i] = (_Float16)whh[i];
  for (int i = tid; i < G4x*2*Hx; i += 512) wih_s[i] = (_Float16)wih[i];
  __syncthreads();

  const int wave = tid >> 5, lane = tid & 31;
  const int bt = wave * 16;
  const int lm = lane & 15;
  const bool hig = lane >= 16;
  _Float16* slab = hsl[wave];

  const int koff  = hig ? 8 : 0;
  const int kbase = hig ? 8 : 0;
  const int mrow  = hig ? 8 : 0;

  v16h whhB[8];
  #pragma unroll
  for (int n = 0; n < 8; ++n)
    whhB[n] = loadB(&whh_s[(n*16 + lm)*Hx + kbase]);

  float biasn[8];
  #pragma unroll
  for (int n = 0; n < 8; ++n){ int g = n*16 + lm; biasn[n] = bih[g] + bhh[g]; }

  v8f cs0, cs1;
  #pragma unroll
  for (int e = 0; e < 8; ++e){ cs0[e] = 0.0f; cs1[e] = 0.0f; }
  v16h ah;
  #pragma unroll
  for (int e = 0; e < 16; ++e) ah[e] = (_Float16)0.0f;

  for (int t = 0; t < Tx; ++t){
    const int tt = dir ? (Tx-1-t) : t;

    // input A-tiles: two K=32 chunks of hs0[b, tt, 0:64]
    const _Float16* hp = hs0 + ((size_t)(bt + lm)*Tx + tt)*(2*Hx);
    v16h a0 = loadA(hp + koff);
    v16h a1 = loadA(hp + 32 + koff);

    v8f z[8];
    #pragma unroll
    for (int n = 0; n < 8; ++n){
      v8f c;
      #pragma unroll
      for (int e = 0; e < 8; ++e) c[e] = biasn[n];
      c = __builtin_amdgcn_wmma_f32_16x16x32_f16(false, ah, false, whhB[n], (short)0, c, false, false);
      int bi = opaque((n*16 + lm)*(2*Hx) + kbase);   // streamed (non-hoistable) LDS tiles
      v16h b0 = loadB(&wih_s[bi]);
      c = __builtin_amdgcn_wmma_f32_16x16x32_f16(false, a0, false, b0, (short)0, c, false, false);
      v16h b1 = loadB(&wih_s[bi + 32]);
      c = __builtin_amdgcn_wmma_f32_16x16x32_f16(false, a1, false, b1, (short)0, c, false, false);
      z[n] = c;
    }

    v8f h0, h1;
    #pragma unroll
    for (int e = 0; e < 8; ++e){
      float i0 = sigf(z[0][e]), f0 = sigf(z[2][e]), g0 = tnhf(z[4][e]), o0 = sigf(z[6][e]);
      cs0[e] = f0*cs0[e] + i0*g0;
      h0[e]  = o0 * tnhf(cs0[e]);
      float i1 = sigf(z[1][e]), f1 = sigf(z[3][e]), g1 = tnhf(z[5][e]), o1 = sigf(z[7][e]);
      cs1[e] = f1*cs1[e] + i1*g1;
      h1[e]  = o1 * tnhf(cs1[e]);
    }

    #pragma unroll
    for (int r = 0; r < 8; ++r){
      slab[(mrow + r)*Hx + lm     ] = (_Float16)h0[r];
      slab[(mrow + r)*Hx + lm + 16] = (_Float16)h1[r];
    }
    ah = loadA(&slab[lm*Hx + koff]);

    if (t == Tx - 1){   // final hidden state -> hn [B, 2H] f32
      #pragma unroll
      for (int r = 0; r < 8; ++r){
        int brow = bt + mrow + r;
        hn[brow*(2*Hx) + dir*Hx + lm     ] = h0[r];
        hn[brow*(2*Hx) + dir*Hx + lm + 16] = h1[r];
      }
    }
  }
}

// ---------------- Classifier head: relu(hn@wc1^T+bc1)@wc2^T+bc2 ----------------
__global__ __launch_bounds__(256)
void cls_head(const float* __restrict__ hn,
              const float* __restrict__ wc1, const float* __restrict__ bc1,
              const float* __restrict__ wc2, const float* __restrict__ bc2,
              float* __restrict__ out)
{
  int b = threadIdx.x;
  if (b >= Bx) return;
  const float* h = hn + b*(2*Hx);
  float hid[32];
  #pragma unroll 4
  for (int j = 0; j < 32; ++j){
    float a = bc1[j];
    for (int k = 0; k < 64; ++k) a += h[k]*wc1[j*64 + k];
    hid[j] = fmaxf(a, 0.0f);
  }
  #pragma unroll
  for (int o = 0; o < 2; ++o){
    float a = bc2[o];
    for (int j = 0; j < 32; ++j) a += hid[j]*wc2[o*32 + j];
    out[b*2 + o] = a;
  }
}

extern "C" void kernel_launch(void* const* d_in, const int* in_sizes, int n_in,
                              void* d_out, int out_size, void* d_ws, size_t ws_size,
                              hipStream_t stream) {
  const float* x        = (const float*)d_in[0];
  const float* wih_l0f  = (const float*)d_in[1];
  const float* whh_l0f  = (const float*)d_in[2];
  const float* bih_l0f  = (const float*)d_in[3];
  const float* bhh_l0f  = (const float*)d_in[4];
  const float* wih_l0b  = (const float*)d_in[5];
  const float* whh_l0b  = (const float*)d_in[6];
  const float* bih_l0b  = (const float*)d_in[7];
  const float* bhh_l0b  = (const float*)d_in[8];
  const float* wih_l1f  = (const float*)d_in[9];
  const float* whh_l1f  = (const float*)d_in[10];
  const float* bih_l1f  = (const float*)d_in[11];
  const float* bhh_l1f  = (const float*)d_in[12];
  const float* wih_l1b  = (const float*)d_in[13];
  const float* whh_l1b  = (const float*)d_in[14];
  const float* bih_l1b  = (const float*)d_in[15];
  const float* bhh_l1b  = (const float*)d_in[16];
  const float* wc1      = (const float*)d_in[17];
  const float* bc1      = (const float*)d_in[18];
  const float* wc2      = (const float*)d_in[19];
  const float* bc2      = (const float*)d_in[20];

  // workspace layout: hs0 [B,T,2H] f16 (16 MB) | hn [B,2H] f32 (64 KB)
  _Float16* hs0 = (_Float16*)d_ws;
  float*    hn  = (float*)((char*)d_ws + (size_t)Bx*Tx*2*Hx*sizeof(_Float16));

  lstm_l0<<<2, 512, 0, stream>>>(x,
      wih_l0f, whh_l0f, bih_l0f, bhh_l0f,
      wih_l0b, whh_l0b, bih_l0b, bhh_l0b, hs0);
  lstm_l1<<<2, 512, 0, stream>>>(hs0,
      wih_l1f, whh_l1f, bih_l1f, bhh_l1f,
      wih_l1b, whh_l1b, bih_l1b, bhh_l1b, hn);
  cls_head<<<1, 256, 0, stream>>>(hn, wc1, bc1, wc2, bc2, (float*)d_out);
}
